// GeometricModel_55396488183983
// MI455X (gfx1250) — matrix operand order repaired and checked by reading, not compiled, not run on hardware
//
#include <hip/hip_runtime.h>
#include <hip/hip_bf16.h>

// ============================================================================
// Fused hyperbolic MLP for MI455X (gfx1250, wave32, WMMA bf16 + TDM).
//   out = softplus(logmap0(mobius_head(tanh_act(fc2(tanh_act(fc1(map(x))))))))
// One block = 8 waves = 128 rows. Weights DMA'd into double-buffered LDS
// slices by the Tensor Data Mover (with hardware bank-conflict padding);
// GEMMs via v_wmma_f32_16x16x32_bf16 with 4-deep software-pipelined B frags.
// Occupancy is LDS-bound (132 KB -> 2 blocks/WGP), so we request the full
// VGPR budget (waves-per-EU = 1) to let the B-fragment pipeline stay live.
// ============================================================================

typedef __attribute__((ext_vector_type(16))) __bf16 v16bf;
typedef __attribute__((ext_vector_type(8)))  float  v8f;
typedef __attribute__((ext_vector_type(4)))  unsigned v4u;
typedef __attribute__((ext_vector_type(8)))  int      v8i;
typedef __attribute__((ext_vector_type(4)))  int      v4i;

union BfFrag { unsigned u[8]; v16bf v; };

#if __has_builtin(__builtin_amdgcn_tensor_load_to_lds)
#define HAVE_TDM 1
#else
#define HAVE_TDM 0
#endif

#if __has_builtin(__builtin_amdgcn_s_wait_tensorcnt)
#define WAIT_TENSOR() __builtin_amdgcn_s_wait_tensorcnt(0)
#else
#define WAIT_TENSOR() asm volatile("s_wait_tensorcnt 0x0" ::: "memory")
#endif

__device__ __forceinline__ unsigned short f2bf(float f) {
  unsigned u = __builtin_bit_cast(unsigned, f);
  u += 0x7fffu + ((u >> 16) & 1u);            // round-to-nearest-even
  return (unsigned short)(u >> 16);
}
__device__ __forceinline__ float bf2f(unsigned short h) {
  unsigned u = (unsigned)h << 16;
  return __builtin_bit_cast(float, u);
}
__device__ __forceinline__ unsigned pk2(float a, float b) {
  return (unsigned)f2bf(a) | ((unsigned)f2bf(b) << 16);
}
__device__ __forceinline__ void up2(unsigned w, float& a, float& b) {
  a = bf2f((unsigned short)(w & 0xffffu));
  b = bf2f((unsigned short)(w >> 16));
}

constexpr int MT   = 128;   // rows per block
constexpr int AROW = 264;   // padded activation row (bf16): 256 + 8, conflict-free
constexpr int WROW = 40;    // padded weight-slice row (bf16): 32 + 8

#if HAVE_TDM
// One 16 KB K-slice (256 rows x 32 bf16, contiguous in d_ws) -> LDS, with the
// TDM inserting 16 B of padding after every 64 B row so LDS gets the padded
// WROW=40 layout directly.  D# bitfields per CDNA5 ISA 8.3/8.4.
// This toolchain exposes the 6-arg builtin:
//   (v4u g0, v8i g1, v4i g2, v4i g3, v8i extra, i32 cpol)
__device__ __forceinline__ void tdm_load_slice(const unsigned short* gsrc,
                                               unsigned lds_byte_addr) {
  unsigned long long ga = (unsigned long long)gsrc;
  v4u g0;
  g0[0] = 1u;                                     // count=1, user descriptor
  g0[1] = lds_byte_addr;                          // lds_addr [63:32]
  g0[2] = (unsigned)(ga & 0xffffffffu);           // global_addr [95:64]
  g0[3] = (unsigned)((ga >> 32) & 0x01ffffffu)    // global_addr [120:96]
        | (2u << 30);                             // type=2 ("image")
  v8i g1;
  g1[0] = (1 << 16)        // data_size = 2 bytes
        | (1 << 20)        // pad_enable
        | (3 << 22)        // pad_interval: after 16 DWORDs (= 64 B = 32 bf16)
        | (3 << 25);       // pad_amount: 4 DWORDs (= 16 B = 8 bf16)
  g1[1] = (8192 & 0xffff) << 16;                  // tensor_dim0 = 8192 (lo16 @48)
  g1[2] = (1 << 16);                              // tensor_dim0 hi=0; tensor_dim1=1 @80
  g1[3] = (8192 << 16);                           // tile_dim0 = 8192 @112
  g1[4] = 1;                                      // tile_dim1 = 1
  g1[5] = 8192;                                   // tensor_dim0_stride
  g1[6] = 0;
  g1[7] = 0;
  v4i z4 = {0, 0, 0, 0};
  v8i z8 = {0, 0, 0, 0, 0, 0, 0, 0};
  __builtin_amdgcn_tensor_load_to_lds(g0, g1, z4, z4, z8, 0);
}
#endif

// --- prep: repack fp32 row-major W[nrows][256] into bf16 [k_tile][nrows][32]
__global__ void pack_w_bf16(const float* __restrict__ W,
                            unsigned short* __restrict__ dst, int nrows) {
  int idx = blockIdx.x * blockDim.x + threadIdx.x;
  if (idx >= nrows * 256) return;
  int n = idx >> 8;
  int k = idx & 255;
  dst[(k >> 5) * (nrows * 32) + n * 32 + (k & 31)] = f2bf(W[idx]);
}

__global__ __launch_bounds__(256, 1) void hnn_fused(
    const float* __restrict__ x,  const float* __restrict__ b1g,
    const float* __restrict__ b2g, const float* __restrict__ bcg,
    const unsigned short* __restrict__ wp, float* __restrict__ out) {
  __shared__ unsigned short Wbuf[2][256 * WROW];   // 40 KB  double-buffered W slice
  __shared__ unsigned short actb[MT * AROW];       // 66 KB  bf16 activations
  __shared__ unsigned short WcS[8 * 16 * WROW];    // 10 KB  classifier weights
  __shared__ float clsS[MT * 20];                  // 10 KB  classifier fp32 out
  __shared__ float preS[MT];                       // per-row tangent scale
  __shared__ float hsS[MT];                        // per-row manifold scale (last)
  __shared__ float xnS[MT];                        // per-row ||h|| (last)
  __shared__ float biasS[512];                     // b1 | b2

  const int t    = threadIdx.x;
  const int lane = t & 31;
  const int wave = t >> 5;
  const int h16  = lane >> 4;      // K-half selector (WMMA A/B layout)
  const int l16  = lane & 15;
  const int R    = wave * 16;      // this wave's first row
  const long blockRow = (long)blockIdx.x * MT;

  const v8f vzero = {0.f, 0.f, 0.f, 0.f, 0.f, 0.f, 0.f, 0.f};

  // ---- stage classifier weights (packed [kt][16][32]) + biases -------------
  {
    const unsigned short* wc = wp + 2 * 65536;
    int e0 = t * 16;                         // 4096 elems / 256 threads
    int kt = e0 >> 9;
    int nn = (e0 >> 5) & 15;
    int kk = e0 & 31;
    uint4 qa = *(const uint4*)(wc + e0);
    uint4 qb = *(const uint4*)(wc + e0 + 8);
    *(uint4*)(&WcS[kt * (16 * WROW) + nn * WROW + kk])     = qa;
    *(uint4*)(&WcS[kt * (16 * WROW) + nn * WROW + kk + 8]) = qb;
    biasS[t]       = b1g[t];
    biasS[256 + t] = b2g[t];
  }

  // ---- stage x tile (fp32 -> bf16), compute pre0 = atanh(min(tanh r,.999))/r
  {
    int row = t >> 1, hf = t & 1;
    const float4* xp4 = (const float4*)(x + (blockRow + row) * 256 + hf * 128);
    uint4* ap4 = (uint4*)(actb + row * AROW + hf * 128);
    float ss = 0.f;
#pragma unroll
    for (int i = 0; i < 16; ++i) {
      float4 a = xp4[2 * i];
      float4 b = xp4[2 * i + 1];
      ss += a.x * a.x + a.y * a.y + a.z * a.z + a.w * a.w
          + b.x * b.x + b.y * b.y + b.z * b.z + b.w * b.w;
      uint4 o;
      o.x = pk2(a.x, a.y); o.y = pk2(a.z, a.w);
      o.z = pk2(b.x, b.y); o.w = pk2(b.z, b.w);
      ap4[i] = o;
    }
    ss += __shfl_xor(ss, 1, 32);
    float r0 = sqrtf(ss);
    float m0 = fminf(tanhf(r0), 0.999f);                 // expmap0 + proj norm
    if (hf == 0)
      preS[row] = atanhf(fminf(m0, 0.99999f)) / fmaxf(r0, 1e-15f);  // logmap0
  }
  __syncthreads();

  // ---- rowwise nonlinearity: t1 -> expmap0/proj -> logmap0 -> tanh -> ... --
  auto rowwise = [&](bool last) {
    int row = t >> 1, hf = t & 1;
    uint4* ap4 = (uint4*)(actb + row * AROW + hf * 128);
    float ss = 0.f;
    for (int i = 0; i < 16; ++i) {
      uint4 q = ap4[i];
      float f0, f1, f2, f3, f4, f5, f6, f7;
      up2(q.x, f0, f1); up2(q.y, f2, f3); up2(q.z, f4, f5); up2(q.w, f6, f7);
      ss += f0*f0 + f1*f1 + f2*f2 + f3*f3 + f4*f4 + f5*f5 + f6*f6 + f7*f7;
    }
    ss += __shfl_xor(ss, 1, 32);
    float r  = sqrtf(ss);
    float m1 = fminf(tanhf(r), 0.999f);
    float g  = atanhf(fminf(m1, 0.99999f)) / fmaxf(r, 1e-15f);  // u = g * t1
    float sv = 0.f;
    for (int i = 0; i < 16; ++i) {
      uint4 q = ap4[i];
      float f0, f1, f2, f3, f4, f5, f6, f7;
      up2(q.x, f0, f1); up2(q.y, f2, f3); up2(q.z, f4, f5); up2(q.w, f6, f7);
      f0 = tanhf(g * f0); f1 = tanhf(g * f1); f2 = tanhf(g * f2); f3 = tanhf(g * f3);
      f4 = tanhf(g * f4); f5 = tanhf(g * f5); f6 = tanhf(g * f6); f7 = tanhf(g * f7);
      sv += f0*f0 + f1*f1 + f2*f2 + f3*f3 + f4*f4 + f5*f5 + f6*f6 + f7*f7;
      uint4 o;
      o.x = pk2(f0, f1); o.y = pk2(f2, f3); o.z = pk2(f4, f5); o.w = pk2(f6, f7);
      ap4[i] = o;
    }
    sv += __shfl_xor(sv, 1, 32);
    float nv = sqrtf(sv);
    float m2 = fminf(tanhf(nv), 0.999f);
    if (hf == 0) {
      if (!last) {
        preS[row] = atanhf(fminf(m2, 0.99999f)) / fmaxf(nv, 1e-15f);
      } else {
        hsS[row] = m2 / fmaxf(nv, 1e-15f);   // manifold point h = hs * v
        xnS[row] = m2;                        // ||h||
      }
    }
  };

#if HAVE_TDM
  const unsigned ldsW0 = (unsigned)(unsigned long long)&Wbuf[0][0];
  const unsigned ldsW1 = (unsigned)(unsigned long long)&Wbuf[1][0];
#endif

  // ============================ dense layers ===============================
  for (int l = 0; l < 2; ++l) {
    const unsigned short* wm = wp + l * 65536;
#if HAVE_TDM
    if (wave == 0) {                 // TDM: DMA slice 0 (EXEC-independent issue)
      tdm_load_slice(wm, ldsW0);
      WAIT_TENSOR();
    }
    __syncthreads();
#else
    {  // fallback: cooperative register staging of slice 0
      const uint4* s4 = (const uint4*)(wm + t * 32);
      uint4* d4 = (uint4*)(&Wbuf[0][t * WROW]);
      d4[0] = s4[0]; d4[1] = s4[1]; d4[2] = s4[2]; d4[3] = s4[3];
    }
    __syncthreads();
#endif

    v8f acc[16];
#pragma unroll
    for (int n = 0; n < 16; ++n) acc[n] = vzero;

    for (int k = 0; k < 8; ++k) {
#if HAVE_TDM
      if (k < 7 && wave == 0)        // overlap next-slice DMA with WMMA
        tdm_load_slice(wm + (k + 1) * 8192, ((k + 1) & 1) ? ldsW1 : ldsW0);
#else
      uint4 nxt0, nxt1, nxt2, nxt3;
      if (k < 7) {
        const uint4* s4 = (const uint4*)(wm + (k + 1) * 8192 + t * 32);
        nxt0 = s4[0]; nxt1 = s4[1]; nxt2 = s4[2]; nxt3 = s4[3];
      }
#endif
      // A fragment: 16x32 bf16 (this wave's rows), contiguous k = 16*h16 + s (same as B)
      BfFrag A;
      const unsigned short* ar = actb + (R + l16) * AROW + k * 32 + h16 * 16;
      *(uint4*)(&A.u[0]) = *(const uint4*)ar;
      *(uint4*)(&A.u[4]) = *(const uint4*)(ar + 8);

      // 4-deep software-pipelined B fragments, load-ahead distance 2:
      // at wmma(n), only loads for n+1 and n+2 are outstanding -> partial waits
      const unsigned short* wbase = &Wbuf[k & 1][0] + h16 * 16;
      BfFrag Bf[4];
#pragma unroll
      for (int p = 0; p < 2; ++p) {
        const unsigned short* br = wbase + (p * 16 + l16) * WROW;
        *(uint4*)(&Bf[p].u[0]) = *(const uint4*)br;
        *(uint4*)(&Bf[p].u[4]) = *(const uint4*)(br + 8);
      }
#pragma unroll
      for (int n = 0; n < 16; ++n) {
        if (n + 2 < 16) {
          const unsigned short* br = wbase + ((n + 2) * 16 + l16) * WROW;
          *(uint4*)(&Bf[(n + 2) & 3].u[0]) = *(const uint4*)br;
          *(uint4*)(&Bf[(n + 2) & 3].u[4]) = *(const uint4*)(br + 8);
        }
        acc[n] = __builtin_amdgcn_wmma_f32_16x16x32_bf16(
            false, A.v, false, Bf[n & 3].v, (short)0, acc[n], false, false);
      }
#if HAVE_TDM
      if (k < 7 && wave == 0) WAIT_TENSOR();
#else
      if (k < 7) {
        uint4* d4 = (uint4*)(&Wbuf[(k + 1) & 1][t * WROW]);
        d4[0] = nxt0; d4[1] = nxt1; d4[2] = nxt2; d4[3] = nxt3;
      }
#endif
      __syncthreads();
    }

    // epilogue: t1 = pre[row] * (base @ W^T) + b   (logmap0 scale commutes)
    {
      const int rbase = R + h16 * 8;
      float pr[8];
#pragma unroll
      for (int j = 0; j < 8; ++j) pr[j] = preS[rbase + j];
      const float* bb = &biasS[l * 256];
#pragma unroll
      for (int n = 0; n < 16; ++n) {
        float bn = bb[n * 16 + l16];
#pragma unroll
        for (int j = 0; j < 8; ++j) {
          float y = acc[n][j] * pr[j] + bn;
          actb[(rbase + j) * AROW + n * 16 + l16] = f2bf(y);
        }
      }
    }
    __syncthreads();
    rowwise(l == 1);
    __syncthreads();
  }

  // ============================ classifier =================================
  {
    v8f cacc = vzero;
    for (int k = 0; k < 8; ++k) {
      BfFrag A;
      const unsigned short* ar = actb + (R + l16) * AROW + k * 32 + h16 * 16;
      *(uint4*)(&A.u[0]) = *(const uint4*)ar;
      *(uint4*)(&A.u[4]) = *(const uint4*)(ar + 8);
      BfFrag B;
      const unsigned short* br = &WcS[k * (16 * WROW) + l16 * WROW + h16 * 16];
      *(uint4*)(&B.u[0]) = *(const uint4*)br;
      *(uint4*)(&B.u[4]) = *(const uint4*)(br + 8);
      cacc = __builtin_amdgcn_wmma_f32_16x16x32_bf16(
          false, A.v, false, B.v, (short)0, cacc, false, false);
    }
    const int rbase = R + h16 * 8;
#pragma unroll
    for (int j = 0; j < 8; ++j) {
      int row = rbase + j;
      clsS[row * 20 + l16] = cacc[j] * hsS[row];   // mx = h @ Wc^T
    }
  }
  __syncthreads();

  // ---- per-row Möbius head + logmap0 + softplus ---------------------------
  if (t < MT) {
    const int row = t;
    float mx[16]; float s2 = 0.f;
#pragma unroll
    for (int i = 0; i < 16; ++i) { mx[i] = clsS[row * 20 + i]; s2 += mx[i] * mx[i]; }
    float mxn = sqrtf(s2);
    float xn  = xnS[row];
    float arg = (mxn / fmaxf(xn, 1e-15f)) * atanhf(fminf(xn, 0.99999f));
    float tn  = tanhf(arg);
    float sc  = tn / fmaxf(mxn, 1e-15f);
    if (s2 == 0.f) sc = 0.f;                       // zero-row guard
    float rn = sc * mxn;
    if (rn > 0.999f) sc *= 0.999f / rn;            // proj
    // hyperbolic bias hb = proj(expmap0(bc))
    float hb[16]; float nb2 = 0.f;
#pragma unroll
    for (int i = 0; i < 16; ++i) { hb[i] = bcg[i]; nb2 += hb[i] * hb[i]; }
    float nb = sqrtf(nb2);
    float eh = fminf(tanhf(nb), 0.999f) / fmaxf(nb, 1e-15f);
    float x2 = 0.f, y2 = 0.f, xy = 0.f;
#pragma unroll
    for (int i = 0; i < 16; ++i) {
      mx[i] *= sc; hb[i] *= eh;
      x2 += mx[i] * mx[i]; y2 += hb[i] * hb[i]; xy += mx[i] * hb[i];
    }
    float c1  = 1.f + 2.f * xy + y2;
    float c2  = 1.f - x2;
    float den = fmaxf(1.f + 2.f * xy + x2 * y2, 1e-15f);
    float o[16]; float no2 = 0.f;
#pragma unroll
    for (int i = 0; i < 16; ++i) {
      o[i] = (c1 * mx[i] + c2 * hb[i]) / den;      // mobius_add
      no2 += o[i] * o[i];
    }
    float no = sqrtf(no2);
    float ps = (no > 0.999f) ? (0.999f / no) : 1.f;  // proj
    float n2 = no * ps;
    float ls = atanhf(fminf(n2, 0.99999f)) / fmaxf(n2, 1e-15f) * ps;  // logmap0
    float4* op = (float4*)(out + (blockRow + row) * 16);
#pragma unroll
    for (int i = 0; i < 4; ++i) {
      float v0 = ls * o[4 * i + 0], v1 = ls * o[4 * i + 1];
      float v2 = ls * o[4 * i + 2], v3 = ls * o[4 * i + 3];
      float s0 = v0 > 20.f ? v0 : log1pf(expf(v0));
      float s1 = v1 > 20.f ? v1 : log1pf(expf(v1));
      float s2o = v2 > 20.f ? v2 : log1pf(expf(v2));
      float s3 = v3 > 20.f ? v3 : log1pf(expf(v3));
      op[i] = make_float4(s0, s1, s2o, s3);
    }
  }
}

extern "C" void kernel_launch(void* const* d_in, const int* in_sizes, int n_in,
                              void* d_out, int out_size, void* d_ws, size_t ws_size,
                              hipStream_t stream) {
  const float* x  = (const float*)d_in[0];
  const float* W1 = (const float*)d_in[1];
  const float* b1 = (const float*)d_in[2];
  const float* W2 = (const float*)d_in[3];
  const float* b2 = (const float*)d_in[4];
  const float* Wc = (const float*)d_in[5];
  const float* bc = (const float*)d_in[6];
  unsigned short* wp = (unsigned short*)d_ws;   // bf16 packed weights (~264 KB)
  float* out = (float*)d_out;
  const int N = in_sizes[0] / 256;

  pack_w_bf16<<<(65536 + 255) / 256, 256, 0, stream>>>(W1, wp, 256);
  pack_w_bf16<<<(65536 + 255) / 256, 256, 0, stream>>>(W2, wp + 65536, 256);
  pack_w_bf16<<<(4096 + 255) / 256, 256, 0, stream>>>(Wc, wp + 131072, 16);
  hnn_fused<<<N / MT, 256, 0, stream>>>(x, b1, b2, bc, wp, out);
}
